// MSDeformableAttention_4226247819539
// MI455X (gfx1250) — matrix operand order repaired and checked by reading, not compiled, not run on hardware
//
#include <hip/hip_runtime.h>
#include <hip/hip_bf16.h>

// ---------------------------------------------------------------------------
// MS-Deformable-Attention forward for MI455X (gfx1250, wave32, WMMA bf16,
// Tensor-Data-Mover LDS staging, double-buffered).
// ---------------------------------------------------------------------------

typedef __attribute__((ext_vector_type(16))) __bf16 v16bf;
typedef __attribute__((ext_vector_type(8)))  float  v8f;
typedef __attribute__((ext_vector_type(4)))  unsigned int v4u;
typedef __attribute__((ext_vector_type(8)))  int    v8i;
typedef __attribute__((ext_vector_type(4)))  int    v4i;

#define EMBED   256
#define HEADS   8
#define LEVELS  4
#define POINTS  4
#define HD      32
#define KDIM    256
#define LQN     21760
#define NVN     21760
#define BATCH   2

// LDS A-panel row stride: 32 bf16 + 1 DWORD hardware pad = 34 halfwords (68B)
#define AROW 34

typedef __attribute__((address_space(3))) __bf16 lds_bf16_t;

// ---------------------------------------------------------------------------
// TDM: DMA one 128x32 bf16 tile (rows of A, K-contiguous) into LDS.
// D# per CDNA5 ISA 8.3-8.6:  group0 = {count/type/addr}, group1 = dims.
// pad_enable + pad_interval=16DW + pad_amount=1DW -> LDS row stride 68B,
// which spreads rows across banks (same effect as manual [34] padding).
// Toolchain: clang-23 6-arg builtin (g0, g1, g2, g3, g4, cpol).
// ---------------------------------------------------------------------------
__device__ __forceinline__ void tdm_load_A(const __bf16* gsrc, const __bf16* lds_dst, int M)
{
    const unsigned lds_off = (unsigned)(unsigned long long)(lds_bf16_t*)lds_dst;
    const unsigned long long ga = (unsigned long long)(size_t)gsrc;
    union { unsigned u[4]; v4u v; } g0;
    union { int i[8]; v8i v; } g1;
    g0.u[0] = 1u;                                   // count=1 (valid), load, no gather
    g0.u[1] = lds_off;                              // lds_addr
    g0.u[2] = (unsigned)(ga & 0xffffffffu);         // global_addr[31:0]
    g0.u[3] = (unsigned)((ga >> 32) & 0x01ffffffu)  // global_addr[56:32]
            | (2u << 30);                           // type=2 ("image")
    g1.i[0] = (1 << 16)    // data_size = 2 bytes (bf16)
            | (1 << 20)    // pad_enable
            | (3 << 22);   // pad_interval: 16 DWORDs (64B) ; pad_amount=0 -> 1 DWORD
    g1.i[1] = (int)((unsigned)KDIM << 16);          // tensor_dim0 = K (low16)
    g1.i[2] = (int)((unsigned)M    << 16);          // tensor_dim0 hi=0 | tensor_dim1 lo
    g1.i[3] = (int)(32u << 16);                     // tensor_dim1 hi=0 | tile_dim0=32
    g1.i[4] = 128;                                  // tile_dim1=128, tile_dim2=0
    g1.i[5] = KDIM;                                 // tensor_dim0_stride[31:0] = K
    g1.i[6] = 0;                                    // stride0 hi | stride1 lo
    g1.i[7] = 0;                                    // stride1 hi
    const v4i z4 = {};
    const v8i z8 = {};
    __builtin_amdgcn_tensor_load_to_lds(g0.v, g1.v, z4, z4, z8, 0);
}

// ---------------------------------------------------------------------------
// f32 -> bf16 conversion (one pass; activations are consumed as bf16 by the
// WMMA GEMMs, halving their streaming traffic).
// ---------------------------------------------------------------------------
__global__ __launch_bounds__(256) void cvt_f32_bf16(
    const float* __restrict__ in, __bf16* __restrict__ out, int n)
{
    const size_t i = ((size_t)blockIdx.x * blockDim.x + threadIdx.x) * 4;
    if (i >= (size_t)n) return;
#pragma unroll
    for (int j = 0; j < 4; ++j) out[i + j] = (__bf16)in[i + j];
}

// ---------------------------------------------------------------------------
// GEMM: C[M,N] = A[M,256](bf16) * W[256,N](f32) + bias[N], fp32 accumulate.
// Block: 256 threads = 8 waves; tile 128(M) x 64(N); K stepped by 32.
// A panel: TDM direct-to-LDS, double buffered (TENSORcnt pipelining).
// Each wave: 4 independent accumulators -> 4 back-to-back v_wmma per step.
// MODE 0: C row-major [M,N]
// MODE 1: value-permute: row r=(b*NV+i), col c=(h*32+d)
//         -> C[((b*8+h)*NV + i)*32 + d]
// ---------------------------------------------------------------------------
template<int MODE>
__global__ __launch_bounds__(256) void gemm_bf16_wmma(
    const __bf16* __restrict__ A, const float* __restrict__ W,
    const float* __restrict__ bias, float* __restrict__ C, int N, int M)
{
    __shared__ __align__(128) __bf16 sA[2][128][AROW]; // TDM-padded A panels
    __shared__ __align__(128) __bf16 sBT[64][32];      // B transposed, 64B rows

    const int t     = threadIdx.x;
    const int lane  = t & 31;
    const int wave  = t >> 5;
    const int mBase = blockIdx.x * 128;
    const int nBase = blockIdx.y * 64;

    v8f acc0 = {}, acc1 = {}, acc2 = {}, acc3 = {};

    const __bf16* Abase = A + (size_t)mBase * KDIM;

    // prologue: kick off DMA of K-step 0 into buffer 0
    if (t < 32) tdm_load_A(Abase, &sA[0][0][0], M);

    for (int kk = 0; kk < KDIM / 32; ++kk) {
        const int k0 = kk * 32;
        // ---- stage B panel 32x64 f32 -> bf16, transposed into sBT[n][k] ----
        {
            const int kr = t >> 3;          // 0..31
            const int nc = (t & 7) * 8;     // 0..56
            const float* src = W + (size_t)(k0 + kr) * N + nBase + nc;
#pragma unroll
            for (int j = 0; j < 8; ++j)
                sBT[nc + j][kr] = (__bf16)src[j];
        }
        // ---- pipeline next A tile, then wait for current one ----
        if (t < 32) {
            if (kk + 1 < KDIM / 32) {
                tdm_load_A(Abase + (kk + 1) * 32, &sA[(kk + 1) & 1][0][0], M);
                __builtin_amdgcn_s_wait_tensorcnt(1);   // current tile done
            } else {
                __builtin_amdgcn_s_wait_tensorcnt(0);
            }
        }
        __syncthreads();

        // ---- build fragments per ISA 7.12.2 layouts ----
        const int half = lane >> 4;            // 0: lanes 0-15, 1: lanes 16-31
        const int m    = (wave << 4) + (lane & 15);
        const __bf16 (*sAc)[AROW] = sA[kk & 1];
        v16bf afrag;
#pragma unroll
        for (int e = 0; e < 16; ++e) {
            // A 16x32: VGPRs hold K pairs {0..7}/{16..23} (+8 for upper lanes)
            const int kA = (e & 7) + ((e >> 3) << 4) + (half << 3);
            afrag[e] = sAc[m][kA];
        }
        v16bf bfrag[4];
#pragma unroll
        for (int nt = 0; nt < 4; ++nt) {
            const int n = (nt << 4) + (lane & 15);
#pragma unroll
            for (int e = 0; e < 16; ++e) {
                // B 32x16: K {0..7}/{16..23} (+8 for upper lanes), matching A
                bfrag[nt][e] = sBT[n][(e & 7) + ((e >> 3) << 4) + (half << 3)];
            }
        }

        acc0 = __builtin_amdgcn_wmma_f32_16x16x32_bf16(
                   false, afrag, false, bfrag[0], (short)0, acc0, false, false);
        acc1 = __builtin_amdgcn_wmma_f32_16x16x32_bf16(
                   false, afrag, false, bfrag[1], (short)0, acc1, false, false);
        acc2 = __builtin_amdgcn_wmma_f32_16x16x32_bf16(
                   false, afrag, false, bfrag[2], (short)0, acc2, false, false);
        acc3 = __builtin_amdgcn_wmma_f32_16x16x32_bf16(
                   false, afrag, false, bfrag[3], (short)0, acc3, false, false);
        __syncthreads();
    }

    // ---- epilogue: C/D layout -> VGPR g: M = g + 8*half, N = lane%16 ----
    const int half = lane >> 4;
    v8f accs[4] = {acc0, acc1, acc2, acc3};
#pragma unroll
    for (int nt = 0; nt < 4; ++nt) {
        const int n  = nBase + (nt << 4) + (lane & 15);
        const float bv = bias[n];
#pragma unroll
        for (int g = 0; g < 8; ++g) {
            const int m = mBase + (wave << 4) + g + (half << 3);
            const float v = accs[nt][g] + bv;
            if (MODE == 0) {
                C[(size_t)m * N + n] = v;
            } else {
                const int b = (m >= NVN) ? 1 : 0;     // M == 2*NVN: no idiv
                const int i = m - b * NVN;
                const int h = n >> 5,  d = n & 31;
                C[(((size_t)(b * HEADS + h) * NVN) + i) * HD + d] = v;
            }
        }
    }
}

// ---------------------------------------------------------------------------
// Softmax over 16 logits per (b,q,h), in place.
// ---------------------------------------------------------------------------
__global__ __launch_bounds__(256) void softmax16(float* __restrict__ attn, int total)
{
    const int i = blockIdx.x * blockDim.x + threadIdx.x;
    if (i >= total) return;
    float* p = attn + (size_t)i * 16;
    float v[16], mx = -3.0e38f;
#pragma unroll
    for (int j = 0; j < 16; ++j) { v[j] = p[j]; mx = fmaxf(mx, v[j]); }
    float s = 0.f;
#pragma unroll
    for (int j = 0; j < 16; ++j) { v[j] = __expf(v[j] - mx); s += v[j]; }
    const float inv = 1.0f / s;
#pragma unroll
    for (int j = 0; j < 16; ++j) p[j] = v[j] * inv;
}

// ---------------------------------------------------------------------------
// Deformable bilinear sampling. One wave32 per (b,q,h); lane = channel d.
// vperm is [B][H][NV][32] f32 so each corner gather is a coalesced 128B load;
// value tensor (44 MB) is L2-resident (192 MB L2). Output stored bf16 so the
// final GEMM streams half the bytes.
// ---------------------------------------------------------------------------
__global__ __launch_bounds__(256) void msda_sample(
    const float* __restrict__ vperm, const float* __restrict__ ref,
    const float* __restrict__ off,   const float* __restrict__ attn,
    __bf16* __restrict__ samp)
{
    const int lane = threadIdx.x & 31;
    const int item = blockIdx.x * 8 + (threadIdx.x >> 5);
    const int total = BATCH * LQN * HEADS;
    if (item >= total) return;

    const int h  = item & (HEADS - 1);
    const int bq = item >> 3;            // b*LQ + q
    const int b  = bq / LQN;

    const int Wl_[LEVELS]    = {128, 64, 32, 16};
    const int Hl_[LEVELS]    = {128, 64, 32, 16};
    const int start_[LEVELS] = {0, 16384, 20480, 21504};

    const float* offp = off  + (size_t)bq * (HEADS * LEVELS * POINTS * 2) + h * (LEVELS * POINTS * 2);
    const float* attp = attn + (size_t)bq * (HEADS * LEVELS * POINTS)     + h * (LEVELS * POINTS);
    const float* refp = ref  + (size_t)bq * (LEVELS * 2);
    const float* vb   = vperm + (size_t)(b * HEADS + h) * NVN * HD + lane;

    float acc = 0.f;
#pragma unroll
    for (int l = 0; l < LEVELS; ++l) {
        const int   Wl = Wl_[l], Hl = Hl_[l];
        const float rx = refp[l * 2 + 0];
        const float ry = refp[l * 2 + 1];
        const float* vl = vb + (size_t)start_[l] * HD;

        auto corner = [&](int ix, int iy) -> float {
            const bool valid = (ix >= 0) & (ix < Wl) & (iy >= 0) & (iy < Hl);
            int cx = ix < 0 ? 0 : (ix > Wl - 1 ? Wl - 1 : ix);
            int cy = iy < 0 ? 0 : (iy > Hl - 1 ? Hl - 1 : iy);
            const float v = vl[(size_t)(cy * Wl + cx) * HD];
            return valid ? v : 0.f;
        };

#pragma unroll
        for (int p = 0; p < POINTS; ++p) {
            // loc = ref + off/[W,H]; x = loc_x*W - 0.5  ==  rx*W + ox - 0.5
            const float x = rx * (float)Wl + offp[(l * POINTS + p) * 2 + 0] - 0.5f;
            const float y = ry * (float)Hl + offp[(l * POINTS + p) * 2 + 1] - 0.5f;
            const float x0f = floorf(x), y0f = floorf(y);
            const float fx = x - x0f, fy = y - y0f;
            const int   x0 = (int)x0f, y0 = (int)y0f;
            const float a  = attp[l * POINTS + p];

            const float s =
                corner(x0,     y0    ) * (1.f - fx) * (1.f - fy) +
                corner(x0 + 1, y0    ) * fx         * (1.f - fy) +
                corner(x0,     y0 + 1) * (1.f - fx) * fy         +
                corner(x0 + 1, y0 + 1) * fx         * fy;
            acc += a * s;
        }
    }
    samp[(size_t)bq * EMBED + h * HD + lane] = (__bf16)acc;
}

// ---------------------------------------------------------------------------
// Launch
// ---------------------------------------------------------------------------
extern "C" void kernel_launch(void* const* d_in, const int* in_sizes, int n_in,
                              void* d_out, int out_size, void* d_ws, size_t ws_size,
                              hipStream_t stream)
{
    const float* query = (const float*)d_in[0];
    const float* refpt = (const float*)d_in[1];
    const float* value = (const float*)d_in[2];
    // d_in[3] spatial_shapes, d_in[4] level_start_index: compile-time constants
    const float* W_off = (const float*)d_in[5];
    const float* b_off = (const float*)d_in[6];
    const float* W_att = (const float*)d_in[7];
    const float* b_att = (const float*)d_in[8];
    const float* W_v   = (const float*)d_in[9];
    const float* b_v   = (const float*)d_in[10];
    const float* W_out = (const float*)d_in[11];
    const float* b_out = (const float*)d_in[12];
    float* out = (float*)d_out;

    // -------- workspace layout (bytes) --------
    const size_t NELT = 11141120ull;                   // 2*21760*256
    char*  wsb   = (char*)d_ws;
    float* vperm = (float*)(wsb);                        // NELT f32
    float* offb  = (float*)(wsb +  NELT * 4);            // NELT f32
    float* attnb = (float*)(wsb +  NELT * 8);            // NELT/2 f32
    __bf16* qbf  = (__bf16*)(wsb + NELT * 10);           // NELT bf16
    __bf16* vbf  = (__bf16*)(wsb + NELT * 12);           // NELT bf16
    __bf16* sampb= (__bf16*)(wsb + NELT * 14);           // NELT bf16

    const int M = BATCH * LQN;                           // 43520 = 340 * 128
    dim3 blk(256);

    // one-shot bf16 conversion of GEMM activations
    cvt_f32_bf16<<<(int)(NELT / (256 * 4)), blk, 0, stream>>>(query, qbf, (int)NELT);
    cvt_f32_bf16<<<(int)(NELT / (256 * 4)), blk, 0, stream>>>(value, vbf, (int)NELT);

    gemm_bf16_wmma<1><<<dim3(M / 128, EMBED / 64), blk, 0, stream>>>(vbf, W_v,   b_v,   vperm, EMBED, M);
    gemm_bf16_wmma<0><<<dim3(M / 128, EMBED / 64), blk, 0, stream>>>(qbf, W_off, b_off, offb,  EMBED, M);
    gemm_bf16_wmma<0><<<dim3(M / 128, 128 / 64),   blk, 0, stream>>>(qbf, W_att, b_att, attnb, 128,   M);

    const int totalQH = BATCH * LQN * HEADS;             // 348160
    softmax16<<<(totalQH + 255) / 256, 256, 0, stream>>>(attnb, totalQH);
    msda_sample<<<(totalQH + 7) / 8, 256, 0, stream>>>(vperm, refpt, offb, attnb, sampb);

    gemm_bf16_wmma<0><<<dim3(M / 128, EMBED / 64), blk, 0, stream>>>(sampb, W_out, b_out, out, EMBED, M);
}
